// EdgeClassifier_43885975830933
// MI455X (gfx1250) — matrix operand order repaired and checked by reading, not compiled, not run on hardware
//
#include <hip/hip_runtime.h>
#include <hip/hip_bf16.h>

// EdgeClassifier: out[e] = sigmoid(relu(concat(z[src],z[dst],attr[e]) @ W1 + b1) @ W2 + b2)
// E = 1e6, EMB = 128, FEAT = 16, HIDDEN = 64. K padded 272 -> 288 (9 x 32 bf16 steps);
// K=272 is a bias row (A supplies 1.0, W1T row 272 holds b1) so C initializes to zero.
//
// MI455X plan: z converted once to bf16 in d_ws. One wave32 owns a 16-edge tile.
// The src/dst row gather runs on the Tensor Data Mover in gather mode (4
// descriptors/tile, 8 indexed 256B rows each) into a per-wave double-buffered
// LDS stage, pipelined one tile ahead (TENSORcnt). W1 is transposed once per
// block into LDS bf16; the MMA chain is 36x v_wmma_f32_16x16x32_bf16 per tile.

constexpr int EMB    = 128;
constexpr int FEAT   = 16;
constexpr int KRAW   = 2 * EMB + FEAT;   // 272
constexpr int KDIM   = 288;              // padded to multiple of 32
constexpr int HID    = 64;
constexpr int KT     = KDIM / 32;        // 9 K-steps
constexpr int WAVES  = 8;
constexpr int TILE_E = 16;
constexpr int BLOCK_E = WAVES * TILE_E;  // 128 edges per block-tile
constexpr int STG_ROWS = 32;             // 16 src + 16 dst rows per tile
constexpr int STG_BYTES = STG_ROWS * EMB * 2;   // 8192 B per buffer

typedef __attribute__((ext_vector_type(16))) __bf16 v16bf;
typedef __attribute__((ext_vector_type(8)))  float  v8f;
typedef __attribute__((ext_vector_type(4)))  unsigned int u32x4;
typedef __attribute__((ext_vector_type(4)))  int i32x4;
typedef __attribute__((ext_vector_type(8)))  int i32x8;

union AFrag {
    v16bf v;
    __bf16 h[16];
    unsigned int u[8];
    uint4 q[2];
};

static __device__ __forceinline__ void tdm_gather8(u32x4 g0, i32x8 g1, i32x4 i03, i32x4 i47) {
#if defined(__clang_major__) && (__clang_major__ >= 23)
    i32x8 zero8 = 0;
    __builtin_amdgcn_tensor_load_to_lds(g0, g1, i03, i47, zero8, 0);
#else
    __builtin_amdgcn_tensor_load_to_lds(g0, g1, i03, i47, 0);
#endif
}

static __device__ __forceinline__ void cvt8(__bf16* d, float4 f0, float4 f1) {
    d[0] = (__bf16)f0.x; d[1] = (__bf16)f0.y; d[2] = (__bf16)f0.z; d[3] = (__bf16)f0.w;
    d[4] = (__bf16)f1.x; d[5] = (__bf16)f1.y; d[6] = (__bf16)f1.z; d[7] = (__bf16)f1.w;
}

// One-shot f32 -> bf16 conversion of the node table into scratch.
__global__ void z_to_bf16_kernel(const float* __restrict__ z,
                                 __bf16* __restrict__ zb, int n) {
    union { __bf16 h[8]; uint4 q; } r;
    const int stride = (int)(gridDim.x * blockDim.x);
    for (int i = (int)(blockIdx.x * blockDim.x + threadIdx.x); i * 8 + 7 < n; i += stride) {
        const float4* p = reinterpret_cast<const float4*>(z + i * 8);
        cvt8(r.h, p[0], p[1]);
        reinterpret_cast<uint4*>(zb)[i] = r.q;
    }
}

// Stage W1 transposed into LDS bf16, with b1 as an extra K-row at k == KRAW.
static __device__ __forceinline__ void stageW1T(__bf16* W1T,
                                                const float* __restrict__ W1,
                                                const float* __restrict__ b1,
                                                int tid, int nthreads) {
    for (int i = tid; i < HID * KDIM; i += nthreads) {
        int n = i / KDIM;
        int k = i - n * KDIM;
        float v = (k < KRAW) ? W1[k * HID + n] : ((k == KRAW) ? b1[n] : 0.f);
        W1T[i] = (__bf16)v;
    }
}

// Fill one half (8 bf16) of an A fragment. kk = kt*32 + half*16 is compile-time.
// aBase: TDM stage row base (bf16, rows 0..15 src, +4096B dst) or null in f32 mode.
template <bool TDM>
static __device__ __forceinline__ void aHalf(AFrag& a, int half, int kk, int hi,
                                             const char* aBase,
                                             const float* __restrict__ srcF,
                                             const float* __restrict__ dstF,
                                             const float* __restrict__ attrP) {
    if (kk < 2 * EMB) {
        if constexpr (TDM) {
            const int offs = (kk < EMB) ? kk * 2 : 4096 + (kk - 2 * EMB) * 2 + 2 * EMB * 2 - 2 * EMB * 2;
            // src rows at [0,4096), dst rows at [4096,8192)
            a.q[half] = *reinterpret_cast<const uint4*>(
                aBase + ((kk < EMB) ? kk * 2 : 4096 + (kk - EMB) * 2));
            (void)offs;
        } else {
            const float4* p = (kk < EMB)
                ? reinterpret_cast<const float4*>(srcF + kk)
                : reinterpret_cast<const float4*>(dstF + (kk - EMB));
            cvt8(a.h + half * 8, p[0], p[1]);
        }
    } else if (kk < KRAW) {
        const float4* p = reinterpret_cast<const float4*>(attrP + (kk - 2 * EMB));
        cvt8(a.h + half * 8, p[0], p[1]);
    } else if (kk == KRAW) {
        // bias row: element K==272 gets 1.0 (hi==0 lanes only), rest zero.
        a.u[half * 4 + 0] = (hi == 0) ? 0x3F80u : 0u;   // bf16 1.0 packed low
        a.u[half * 4 + 1] = 0u;
        a.u[half * 4 + 2] = 0u;
        a.u[half * 4 + 3] = 0u;
    } else {
        a.q[half] = make_uint4(0u, 0u, 0u, 0u);
    }
}

// ---------------- TDM-gather main kernel (bf16 z in d_ws) ----------------
__global__ void EdgeClassifier_tdm_kernel(const __bf16* __restrict__ zbf,
                                          const long long* __restrict__ eidx,
                                          const float* __restrict__ attr,
                                          const float* __restrict__ W1,
                                          const float* __restrict__ b1,
                                          const float* __restrict__ W2,
                                          const float* __restrict__ b2,
                                          float* __restrict__ out,
                                          int E, int numTiles, int nNodes) {
    __shared__ __align__(16) __bf16 W1T[HID * KDIM];
    __shared__ __align__(16) __bf16 stage[WAVES][2][STG_ROWS * EMB];

    stageW1T(W1T, W1, b1, threadIdx.x, blockDim.x);
    __syncthreads();

    const int tid  = threadIdx.x;
    const int lane = tid & 31;
    const int wave = tid >> 5;
    const int lo   = lane & 15;
    const int hi   = lane >> 4;

    const float b2s = b2[0];
    float w2v[4];
#pragma unroll
    for (int t = 0; t < 4; ++t) w2v[t] = W2[t * 16 + lo];

    // Single LDS base for all 72 B-fragment loads (16-bit DS immediates).
    const char* wbase = reinterpret_cast<const char*>(&W1T[lo * KDIM]) + hi * 16;

    // ---- Loop-invariant Tensor DMA descriptor pieces (gather mode) ----
    i32x8 g1;
    g1[0] = (1 << 16);                                        // data_size = 2 bytes
    g1[1] = (int)((EMB & 0xFFFF) << 16);                      // tensor_dim0[15:0]
    g1[2] = (int)((nNodes & 0xFFFF) << 16);                   // dim0 hi=0, tensor_dim1[15:0]
    g1[3] = (int)(((nNodes >> 16) & 0xFFFF) | (EMB << 16));   // tensor_dim1[31:16], tile_dim0
    g1[4] = 8;                                                // tile_dim1 = #indices
    g1[5] = EMB;                                              // tensor_dim0_stride
    g1[6] = 0;
    g1[7] = 0;

    const unsigned long long ga = (unsigned long long)(const void*)zbf;
    const unsigned g0w0 = 1u | (1u << 30) | (1u << 31);       // count=1, 32b idx, gather
    const unsigned g0w2 = (unsigned)ga;
    const unsigned g0w3 = (unsigned)((ga >> 32) & 0x01FFFFFFull) | (2u << 30);  // type=2
    const unsigned ldsWave = (unsigned)(size_t)(void*)&stage[wave][0][0];

    const char* aWave = reinterpret_cast<const char*>(&stage[wave][0][0]) + lo * 256 + hi * 16;

    const long long eBase = (long long)hi * (long long)E;
    const int stride = (int)gridDim.x;

    auto issueTile = [&](int buf, int idxVal) {
        const unsigned ldsBase = ldsWave + (unsigned)buf * (unsigned)STG_BYTES;
#pragma unroll
        for (int d = 0; d < 4; ++d) {
            u32x4 g0;
            g0[0] = g0w0;
            g0[1] = ldsBase + (unsigned)d * 2048u;            // 8 rows x 256B per desc
            g0[2] = g0w2;
            g0[3] = g0w3;
            i32x4 i03, i47;
            i03[0] = __builtin_amdgcn_readlane(idxVal, d * 8 + 0);
            i03[1] = __builtin_amdgcn_readlane(idxVal, d * 8 + 1);
            i03[2] = __builtin_amdgcn_readlane(idxVal, d * 8 + 2);
            i03[3] = __builtin_amdgcn_readlane(idxVal, d * 8 + 3);
            i47[0] = __builtin_amdgcn_readlane(idxVal, d * 8 + 4);
            i47[1] = __builtin_amdgcn_readlane(idxVal, d * 8 + 5);
            i47[2] = __builtin_amdgcn_readlane(idxVal, d * 8 + 6);
            i47[3] = __builtin_amdgcn_readlane(idxVal, d * 8 + 7);
            tdm_gather8(g0, g1, i03, i47);
        }
    };

    int tile = (int)blockIdx.x;
    int buf  = 0;

    {   // Prologue: kick off the gather for the first tile.
        const int e0 = tile * BLOCK_E + wave * TILE_E;
        if (tile < numTiles && e0 < E) {
            int eM = e0 + lo; if (eM >= E) eM = E - 1;
            issueTile(0, (int)eidx[eBase + eM]);
        }
    }

    for (; tile < numTiles; tile += stride) {
        const int e0 = tile * BLOCK_E + wave * TILE_E;
        const bool ok = (e0 < E);

        const int tileN = tile + stride;
        bool okN = false;
        if (tileN < numTiles) {
            const int e0n = tileN * BLOCK_E + wave * TILE_E;
            okN = (e0n < E);
            if (okN) {
                int eMn = e0n + lo; if (eMn >= E) eMn = E - 1;
                const int idxN = (int)eidx[eBase + eMn];
                asm volatile("s_wait_dscnt 0" ::: "memory");   // prior reads of buf^1 done
                issueTile(buf ^ 1, idxN);
            }
        }

        if (ok) {
            if (okN) __builtin_amdgcn_s_wait_tensorcnt(4);
            else     __builtin_amdgcn_s_wait_tensorcnt(0);
            asm volatile("" ::: "memory");

            int eMine = e0 + lo; if (eMine >= E) eMine = E - 1;
            const float* attrP = attr + (long long)eMine * FEAT + hi * 8;
            const char*  aBase = aWave + buf * STG_BYTES;

            v8f c[4] = {};          // zero-init: bias comes from the K=272 row

#pragma unroll
            for (int kt = 0; kt < KT; ++kt) {
                AFrag a;
#pragma unroll
                for (int half = 0; half < 2; ++half) {
                    aHalf<true>(a, half, kt * 32 + half * 16, hi, aBase, nullptr, nullptr, attrP);
                }
#pragma unroll
                for (int t = 0; t < 4; ++t) {
                    AFrag b;
                    b.q[0] = *reinterpret_cast<const uint4*>(wbase + t * 9216 + kt * 64);
                    b.q[1] = *reinterpret_cast<const uint4*>(wbase + t * 9216 + kt * 64 + 32);
                    c[t] = __builtin_amdgcn_wmma_f32_16x16x32_bf16(
                        false, a.v, false, b.v, (short)0, c[t], false, false);
                }
            }

            float acc[8];
#pragma unroll
            for (int j = 0; j < 8; ++j) acc[j] = 0.f;
#pragma unroll
            for (int t = 0; t < 4; ++t) {
#pragma unroll
                for (int j = 0; j < 8; ++j) {
                    float hv = c[t][j];
                    hv = hv > 0.f ? hv : 0.f;
                    acc[j] += hv * w2v[t];
                }
            }
#pragma unroll
            for (int j = 0; j < 8; ++j) {
                acc[j] += __shfl_xor(acc[j], 8, 32);
                acc[j] += __shfl_xor(acc[j], 4, 32);
                acc[j] += __shfl_xor(acc[j], 2, 32);
                acc[j] += __shfl_xor(acc[j], 1, 32);
            }
            if (lo == 0) {
                const int rbase = e0 + hi * 8;
#pragma unroll
                for (int j = 0; j < 8; ++j) {
                    const int e = rbase + j;
                    if (e < E) out[e] = 1.f / (1.f + __expf(-(acc[j] + b2s)));
                }
            }
        }
        buf ^= 1;
    }
}

// ---------------- Fallback: direct f32 gather, no scratch needed ----------------
__global__ void EdgeClassifier_f32_kernel(const float* __restrict__ z,
                                          const long long* __restrict__ eidx,
                                          const float* __restrict__ attr,
                                          const float* __restrict__ W1,
                                          const float* __restrict__ b1,
                                          const float* __restrict__ W2,
                                          const float* __restrict__ b2,
                                          float* __restrict__ out,
                                          int E, int numTiles) {
    __shared__ __align__(16) __bf16 W1T[HID * KDIM];
    stageW1T(W1T, W1, b1, threadIdx.x, blockDim.x);
    __syncthreads();

    const int tid  = threadIdx.x;
    const int lane = tid & 31;
    const int wave = tid >> 5;
    const int lo   = lane & 15;
    const int hi   = lane >> 4;
    const float b2s = b2[0];
    float w2v[4];
#pragma unroll
    for (int t = 0; t < 4; ++t) w2v[t] = W2[t * 16 + lo];
    const char* wbase = reinterpret_cast<const char*>(&W1T[lo * KDIM]) + hi * 16;

    for (int tile = blockIdx.x; tile < numTiles; tile += gridDim.x) {
        const int e0 = tile * BLOCK_E + wave * TILE_E;
        if (e0 >= E) continue;
        int eMine = e0 + lo; if (eMine >= E) eMine = E - 1;
        const int myIdx  = (int)eidx[(long long)hi * E + eMine];
        const int srcRow = __shfl(myIdx, lo, 32);
        const int dstRow = __shfl(myIdx, lo + 16, 32);
        const float* srcF  = z + (long long)srcRow * EMB + hi * 8;
        const float* dstF  = z + (long long)dstRow * EMB + hi * 8;
        const float* attrP = attr + (long long)eMine * FEAT + hi * 8;

        v8f c[4] = {};
#pragma unroll
        for (int kt = 0; kt < KT; ++kt) {
            AFrag a;
#pragma unroll
            for (int half = 0; half < 2; ++half) {
                aHalf<false>(a, half, kt * 32 + half * 16, hi, nullptr, srcF, dstF, attrP);
            }
#pragma unroll
            for (int t = 0; t < 4; ++t) {
                AFrag b;
                b.q[0] = *reinterpret_cast<const uint4*>(wbase + t * 9216 + kt * 64);
                b.q[1] = *reinterpret_cast<const uint4*>(wbase + t * 9216 + kt * 64 + 32);
                c[t] = __builtin_amdgcn_wmma_f32_16x16x32_bf16(
                    false, a.v, false, b.v, (short)0, c[t], false, false);
            }
        }
        float acc[8];
#pragma unroll
        for (int j = 0; j < 8; ++j) acc[j] = 0.f;
#pragma unroll
        for (int t = 0; t < 4; ++t) {
#pragma unroll
            for (int j = 0; j < 8; ++j) {
                float hv = c[t][j];
                hv = hv > 0.f ? hv : 0.f;
                acc[j] += hv * w2v[t];
            }
        }
#pragma unroll
        for (int j = 0; j < 8; ++j) {
            acc[j] += __shfl_xor(acc[j], 8, 32);
            acc[j] += __shfl_xor(acc[j], 4, 32);
            acc[j] += __shfl_xor(acc[j], 2, 32);
            acc[j] += __shfl_xor(acc[j], 1, 32);
        }
        if (lo == 0) {
            const int rbase = e0 + hi * 8;
#pragma unroll
            for (int j = 0; j < 8; ++j) {
                const int e = rbase + j;
                if (e < E) out[e] = 1.f / (1.f + __expf(-(acc[j] + b2s)));
            }
        }
    }
}

extern "C" void kernel_launch(void* const* d_in, const int* in_sizes, int n_in,
                              void* d_out, int out_size, void* d_ws, size_t ws_size,
                              hipStream_t stream) {
    const float*     z    = (const float*)d_in[0];
    const long long* eidx = (const long long*)d_in[1];   // int64 [2, E]
    const float*     attr = (const float*)d_in[2];
    const float*     W1   = (const float*)d_in[3];
    const float*     b1   = (const float*)d_in[4];
    const float*     W2   = (const float*)d_in[5];
    const float*     b2   = (const float*)d_in[6];
    float*           out  = (float*)d_out;

    const int nZ       = in_sizes[0];
    const int nNodes   = nZ / EMB;
    const int E        = in_sizes[1] / 2;
    const int numTiles = (E + BLOCK_E - 1) / BLOCK_E;
    int grid = numTiles < 2048 ? numTiles : 2048;

    const bool useBf = (ws_size >= (size_t)nZ * sizeof(unsigned short)) && (nZ % 8 == 0);
    if (useBf) {
        __bf16* zb = (__bf16*)d_ws;
        int cgrid = (nZ / 8 + 255) / 256;
        if (cgrid > 4096) cgrid = 4096;
        z_to_bf16_kernel<<<cgrid, 256, 0, stream>>>(z, zb, nZ);
        EdgeClassifier_tdm_kernel<<<grid, WAVES * 32, 0, stream>>>(
            zb, eidx, attr, W1, b1, W2, b2, out, E, numTiles, nNodes);
    } else {
        EdgeClassifier_f32_kernel<<<grid, WAVES * 32, 0, stream>>>(
            z, eidx, attr, W1, b1, W2, b2, out, E, numTiles);
    }
}
